// GraphFeatureExtractor_38585986187615
// MI455X (gfx1250) — matrix operand order repaired and checked by reading, not compiled, not run on hardware
//
#include <hip/hip_runtime.h>
#include <hip/hip_bf16.h>

typedef __attribute__((ext_vector_type(16))) _Float16 v16h;
typedef __attribute__((ext_vector_type(8)))  float    v8f;

#define FDIM 128
#define BN_EPS 1e-5f

// ---------------------------------------------------------------------------
// agg = h  (vectorized copy; GIN eps=0 means agg starts as self-contribution)
// ---------------------------------------------------------------------------
__global__ void copy_f4(const float4* __restrict__ in, float4* __restrict__ out, int n4) {
  int i = blockIdx.x * blockDim.x + threadIdx.x;
  if (i < n4) out[i] = in[i];
}

// ---------------------------------------------------------------------------
// agg[dst] += h[src]  — one thread per (edge, 4-feature chunk)
// ---------------------------------------------------------------------------
__global__ void scatter_add(const float* __restrict__ h, const int* __restrict__ src,
                            const int* __restrict__ dst, float* __restrict__ agg, int E) {
  int idx = blockIdx.x * blockDim.x + threadIdx.x;
  int e = idx >> 5;
  if (e >= E) return;
  int c = (idx & 31) << 2;
  int s = src[e], d = dst[e];
  const float4 v = *(const float4*)(h + (size_t)s * FDIM + c);
  float* o = agg + (size_t)d * FDIM + c;
  atomicAdd(o + 0, v.x);
  atomicAdd(o + 1, v.y);
  atomicAdd(o + 2, v.z);
  atomicAdd(o + 3, v.w);
}

// ---------------------------------------------------------------------------
// Pre-convert W1/W2 (f32, [L][K][N]) -> transposed f16 Wt[mat][n*128+k]
// so GEMM B-operand loads are contiguous f16 pairs.
// ---------------------------------------------------------------------------
__global__ void convert_wt(const float* __restrict__ W1, const float* __restrict__ W2,
                           _Float16* __restrict__ Wt, int total) {
  int idx = blockIdx.x * blockDim.x + threadIdx.x;
  if (idx >= total) return;
  int mat   = idx >> 14;            // / 16384  (0..5)
  int l     = mat >> 1;
  const float* W = (mat & 1) ? W2 : W1;
  int local = idx & 16383;
  int n = local >> 7;               // output column
  int k = local & 127;              // reduction index
  Wt[idx] = (_Float16)W[(size_t)l * 16384 + (size_t)k * 128 + n];
}

// ---------------------------------------------------------------------------
// Y = relu(X @ W + b) using v_wmma_f32_16x16x32_f16.
// Block = 256 threads = 8 waves; block computes rows [16*bx,16*bx+16),
// wave w computes cols [16*w,16*w+16). K=128 -> 4 WMMA steps.
// Register layouts per CDNA5 ISA 7.12.2:
//   A 16x32 f16: lane<16 -> M=lane, VGPR v holds K pair; lanes 16-31 get K+8/K+24 halves.
//   B 32x16 f16: lanes 0-15 -> N=lane, K=0..7,16..23; lanes 16-31 -> N=lane-16, K=8..15,24..31.
//   C/D f32:     VGPR r -> M = r + 8*(lane>=16), N = lane%16.
// ---------------------------------------------------------------------------
__global__ void __launch_bounds__(256)
gemm_relu_wmma(const float* __restrict__ X, const _Float16* __restrict__ Wt,
               const float* __restrict__ bias, float* __restrict__ Y) {
  const int wave = threadIdx.x >> 5;
  const int lane = threadIdx.x & 31;
  const int half = lane >> 4;
  const int mn   = lane & 15;
  const int row0 = blockIdx.x * 16;
  const int col0 = wave * 16;

  v8f acc = {};
  #pragma unroll
  for (int kb = 0; kb < FDIM; kb += 32) {
    v16h a, b;
    #pragma unroll
    for (int v = 0; v < 8; ++v) {
      // A-matrix K position for this VGPR (within the 32-wide K window)
      const int kA = ((v & 4) ? 16 : 0) + ((v & 3) * 2) + half * 8;
      const float* ap = X + (size_t)(row0 + mn) * FDIM + kb + kA;
      a[2 * v]     = (_Float16)ap[0];
      a[2 * v + 1] = (_Float16)ap[1];
      // B-matrix K position (Wt is [n][k] so this is a contiguous f16 pair)
      const int kB = ((v & 4) ? 16 : 0) + ((v & 3) * 2) + half * 8;
      const _Float16* bp = Wt + (size_t)(col0 + mn) * FDIM + kb + kB;
      b[2 * v]     = bp[0];
      b[2 * v + 1] = bp[1];
    }
    acc = __builtin_amdgcn_wmma_f32_16x16x32_f16(
        /*neg_a=*/false, a, /*neg_b=*/false, b,
        /*c_mod=*/(short)0, acc, /*reuse_a=*/false, /*reuse_b=*/false);
  }

  const float bv = bias[col0 + mn];
  #pragma unroll
  for (int r = 0; r < 8; ++r) {
    const int row = row0 + r + half * 8;
    const float val = acc[r] + bv;
    Y[(size_t)row * FDIM + col0 + mn] = fmaxf(val, 0.0f);
  }
}

// ---------------------------------------------------------------------------
// BatchNorm statistics: per-feature sum & sumsq -> stats[0..127], stats[128..255]
// ---------------------------------------------------------------------------
__global__ void zero_stats(float* stats) {
  stats[threadIdx.x] = 0.0f;
}

__global__ void __launch_bounds__(256)
bn_stats(const float* __restrict__ h, float* __restrict__ stats, int n_rows) {
  __shared__ float ls[FDIM], lq[FDIM];
  const int f = threadIdx.x & (FDIM - 1);
  if (threadIdx.x < FDIM) { ls[threadIdx.x] = 0.0f; lq[threadIdx.x] = 0.0f; }
  __syncthreads();

  const int rpb    = blockDim.x / FDIM;           // rows covered per block per iter
  const int stride = gridDim.x * rpb;
  float s = 0.0f, q = 0.0f;
  for (int row = blockIdx.x * rpb + (threadIdx.x >> 7); row < n_rows; row += stride) {
    const float v = h[(size_t)row * FDIM + f];
    s += v;
    q += v * v;
  }
  atomicAdd(&ls[f], s);
  atomicAdd(&lq[f], q);
  __syncthreads();
  if (threadIdx.x < FDIM) {
    atomicAdd(&stats[threadIdx.x], ls[threadIdx.x]);
    atomicAdd(&stats[FDIM + threadIdx.x], lq[threadIdx.x]);
  }
}

// ---------------------------------------------------------------------------
// out = gamma * (h - mu) * rsqrt(var + eps) + beta   (biased variance)
// ---------------------------------------------------------------------------
__global__ void bn_apply(const float* __restrict__ h, const float* __restrict__ stats,
                         const float* __restrict__ gamma, const float* __restrict__ beta,
                         float* __restrict__ out, int total, float invN) {
  int idx = blockIdx.x * blockDim.x + threadIdx.x;
  if (idx >= total) return;
  const int f = idx & (FDIM - 1);
  const float mu  = stats[f] * invN;
  const float var = stats[FDIM + f] * invN - mu * mu;
  const float sc  = gamma[f] * rsqrtf(var + BN_EPS);
  out[idx] = sc * (h[idx] - mu) + beta[f];
}

__global__ void copy_batch(const int* __restrict__ batch, int* __restrict__ out, int n) {
  int i = blockIdx.x * blockDim.x + threadIdx.x;
  if (i < n) out[i] = batch[i];
}

// ---------------------------------------------------------------------------
extern "C" void kernel_launch(void* const* d_in, const int* in_sizes, int n_in,
                              void* d_out, int out_size, void* d_ws, size_t ws_size,
                              hipStream_t stream) {
  const float* x     = (const float*)d_in[0];
  const int*   ei    = (const int*)d_in[1];
  const int*   batch = (const int*)d_in[2];
  const float* W1    = (const float*)d_in[3];
  const float* b1    = (const float*)d_in[4];
  const float* W2    = (const float*)d_in[5];
  const float* b2    = (const float*)d_in[6];
  const float* gamma = (const float*)d_in[7];
  const float* beta  = (const float*)d_in[8];

  const int n_nodes = in_sizes[0] / FDIM;         // 50000 (multiple of 16)
  const int E       = in_sizes[1] / 2;            // 800000
  const int L       = in_sizes[3] / (FDIM * FDIM);// 3

  const int* src = ei;
  const int* dst = ei + E;

  const size_t matBytes = (size_t)n_nodes * FDIM * sizeof(float);
  char* ws = (char*)d_ws;
  float*    buf_h = (float*)(ws);
  float*    buf_a = (float*)(ws + matBytes);
  float*    buf_t = (float*)(ws + 2 * matBytes);
  float*    stats = (float*)(ws + 3 * matBytes);
  _Float16* Wt    = (_Float16*)(ws + 3 * matBytes + 1024);

  // one-time per launch: f32 -> f16 transposed weights
  {
    const int total = L * 2 * FDIM * FDIM;
    convert_wt<<<(total + 255) / 256, 256, 0, stream>>>(W1, W2, Wt, total);
  }

  const int n4    = n_nodes * FDIM / 4;
  const int total = n_nodes * FDIM;
  const float invN = 1.0f / (float)n_nodes;

  for (int l = 0; l < L; ++l) {
    const float* h_in = (l == 0) ? x : buf_h;

    // agg = h_in ; agg[dst] += h_in[src]
    copy_f4<<<(n4 + 255) / 256, 256, 0, stream>>>((const float4*)h_in, (float4*)buf_a, n4);
    {
      const long long thr = (long long)E * 32;
      scatter_add<<<(int)((thr + 255) / 256), 256, 0, stream>>>(h_in, src, dst, buf_a, E);
    }

    // MLP: relu(agg @ W1 + b1) then relu(... @ W2 + b2)
    gemm_relu_wmma<<<n_nodes / 16, 256, 0, stream>>>(
        buf_a, Wt + (size_t)(2 * l) * FDIM * FDIM, b1 + l * FDIM, buf_t);
    gemm_relu_wmma<<<n_nodes / 16, 256, 0, stream>>>(
        buf_t, Wt + (size_t)(2 * l + 1) * FDIM * FDIM, b2 + l * FDIM, buf_a);

    // BatchNorm (training-mode batch stats)
    zero_stats<<<1, 256, 0, stream>>>(stats);
    bn_stats<<<256, 256, 0, stream>>>(buf_a, stats, n_nodes);
    float* out_h = (l == L - 1) ? (float*)d_out : buf_h;
    bn_apply<<<(total + 255) / 256, 256, 0, stream>>>(
        buf_a, stats, gamma + l * FDIM, beta + l * FDIM, out_h, total, invN);
  }

  // second tuple element: batch indices appended after h in d_out
  copy_batch<<<(n_nodes + 255) / 256, 256, 0, stream>>>(
      batch, (int*)((float*)d_out + (size_t)n_nodes * FDIM), n_nodes);
}
